// HyperbolicMultiheadAttention_9431748182624
// MI455X (gfx1250) — matrix operand order repaired and checked by reading, not compiled, not run on hardware
//
#include <hip/hip_runtime.h>
#include <hip/hip_bf16.h>

// ---------------------------------------------------------------------------
// CDNA5 / gfx1250 bf16 WMMA implementation of hyperbolic multihead attention.
// All matmuls use v_wmma_f32_16x16x32_bf16 (wave32, 16x16 tiles, f32 accum).
// Key identity: exp(-acosh(s)) = 1/(s+sqrt(s^2-1)) = s - sqrt(s^2-1) for s>=1
// (product of the two forms is exactly 1) -> one raw v_sqrt_f32, no division.
// Attention Q/V tiles are staged with gfx1250 async global->LDS DMA
// (GLOBAL_LOAD_ASYNC_TO_LDS_B128, ASYNCcnt path); the K tile needs a
// transpose so it stays on the VGPR path and overlaps with the DMA.
// ---------------------------------------------------------------------------

typedef __attribute__((ext_vector_type(16))) __bf16 v16bf;
typedef __attribute__((ext_vector_type(8)))  float  v8f;
typedef __attribute__((__vector_size__(4 * sizeof(int)))) int v4i_vs;  // builtin's pointee type

#define AS1 __attribute__((address_space(1)))
#define AS3 __attribute__((address_space(3)))

#if defined(__has_builtin)
#if __has_builtin(__builtin_amdgcn_global_load_async_to_lds_b128)
#define HAVE_ASYNC_LDS 1
#endif
#endif
#ifndef HAVE_ASYNC_LDS
#define HAVE_ASYNC_LDS 0
#endif

__device__ __forceinline__ AS1 v4i_vs* as_global_v4(const void* p) {
    return (AS1 v4i_vs*)(unsigned long long)(uintptr_t)p;
}
__device__ __forceinline__ AS3 v4i_vs* as_lds_v4(void* p) {
    // generic "shared" addresses carry the LDS byte offset in their low bits
    return (AS3 v4i_vs*)(unsigned)(uintptr_t)p;
}
__device__ __forceinline__ void wait_async0() {
#if defined(__has_builtin)
#if __has_builtin(__builtin_amdgcn_s_wait_asynccnt)
    __builtin_amdgcn_s_wait_asynccnt(0);
#else
    asm volatile("s_wait_asynccnt 0x0" ::: "memory");
#endif
#endif
}

union Frag {               // one wave32 A or B operand: 16 bf16 = 8 VGPRs
    v16bf v;
    uint4 u[2];
};

__device__ __forceinline__ unsigned short f2bf(float f) {
    union { float f; unsigned u; } x;
    x.f = f;
    unsigned u = x.u;
    u += 0x7FFFu + ((u >> 16) & 1u);   // round-to-nearest-even
    return (unsigned short)(u >> 16);
}

__device__ __forceinline__ v8f wmma_bf16(const Frag& a, const Frag& b, v8f c) {
    return __builtin_amdgcn_wmma_f32_16x16x32_bf16(
        /*neg_a=*/false, a.v, /*neg_b=*/false, b.v,
        /*c_mod=*/(short)0, c, /*reuse_a=*/false, /*reuse_b=*/false);
}

// ---------------------------------------------------------------------------
// GEMM: out[m,n] = sum_k A[m,k] * W[n,k] + bias[n]
// A: MxK fp32 row-major, W: NxK fp32 row-major (both K-contiguous).
// Block tile 128x128, K-step 32, 256 threads = 8 waves in a 4x2 grid,
// each wave owns a 32x64 sub-tile (2x4 WMMA accumulators).
// fp32->bf16 conversion happens while staging to LDS (can't be DMA'd).
// ---------------------------------------------------------------------------
template <bool OUT_BF16>
__global__ __launch_bounds__(256) void gemm_bias_kernel(
    const float* __restrict__ A, const float* __restrict__ W,
    const float* __restrict__ bias, void* __restrict__ out,
    int M, int N, int K)
{
    __shared__ unsigned short Ash[128][32];   // [m][k] bf16
    __shared__ unsigned short Bsh[32][128];   // [k][n] bf16 (W staged transposed)

    const int t    = threadIdx.x;
    const int lane = t & 31;
    const int wv   = t >> 5;
    const int wm   = wv >> 1;        // 0..3 -> 32-row strip
    const int wn   = wv & 1;         // 0..1 -> 64-col strip
    const int m0   = blockIdx.y * 128;
    const int n0   = blockIdx.x * 128;

    v8f acc[2][4];
    for (int i = 0; i < 2; i++)
        for (int j = 0; j < 4; j++)
            for (int e = 0; e < 8; e++) acc[i][j][e] = 0.0f;

    const int half = lane >> 4;      // ISA 16-bit A layout: lanes 16-31 hold K+8
    const int row0 = lane & 15;

    for (int k0 = 0; k0 < K; k0 += 32) {
        // ---- stage A tile (fp32 -> bf16), thread t covers one 16-elem chunk
        {
            const int m  = t >> 1;
            const int kc = (t & 1) * 16;
            const float* gp = A + (size_t)(m0 + m) * K + k0 + kc;
            if (k0 + 32 < K) __builtin_prefetch(gp + 32, 0, 1);  // global_prefetch_b8
            const float4* g = (const float4*)gp;
            float4 f0 = g[0], f1 = g[1], f2 = g[2], f3 = g[3];
            unsigned short* d = &Ash[m][kc];
            d[0]  = f2bf(f0.x); d[1]  = f2bf(f0.y); d[2]  = f2bf(f0.z); d[3]  = f2bf(f0.w);
            d[4]  = f2bf(f1.x); d[5]  = f2bf(f1.y); d[6]  = f2bf(f1.z); d[7]  = f2bf(f1.w);
            d[8]  = f2bf(f2.x); d[9]  = f2bf(f2.y); d[10] = f2bf(f2.z); d[11] = f2bf(f2.w);
            d[12] = f2bf(f3.x); d[13] = f2bf(f3.y); d[14] = f2bf(f3.z); d[15] = f2bf(f3.w);
        }
        // ---- stage W tile transposed: Bsh[k][n] = W[n0+n][k0+k]
        {
            const int n  = t >> 1;
            const int kc = (t & 1) * 16;
            const float4* g = (const float4*)(W + (size_t)(n0 + n) * K + k0 + kc);
            float4 f[4] = { g[0], g[1], g[2], g[3] };
            const float* fs = (const float*)f;
            #pragma unroll
            for (int i = 0; i < 16; i++) Bsh[kc + i][n] = f2bf(fs[i]);
        }
        __syncthreads();

        // ---- A fragments: lane holds row M=row0, K chunks {8*half, 16+8*half}
        Frag afr[2];
        #pragma unroll
        for (int mi = 0; mi < 2; mi++) {
            const int r = wm * 32 + mi * 16 + row0;
            afr[mi].u[0] = *(const uint4*)&Ash[r][8 * half];
            afr[mi].u[1] = *(const uint4*)&Ash[r][16 + 8 * half];
        }
        // ---- B fragments: lane L holds N=L%16, K in table order, read down a Bsh column
        #pragma unroll
        for (int ni = 0; ni < 4; ni++) {
            Frag bfr;
            const int c = wn * 64 + ni * 16;
            #pragma unroll
            for (int e = 0; e < 16; e++)
                ((unsigned short*)&bfr)[e] = Bsh[(e & 7) + 8 * half + 16 * (e >> 3)][c + row0];
            #pragma unroll
            for (int mi = 0; mi < 2; mi++)
                acc[mi][ni] = wmma_bf16(afr[mi], bfr, acc[mi][ni]);
        }
        __syncthreads();
    }

    // ---- epilogue: C layout -> lane L: VGPR r is (M = r + 8*half, N = row0)
    #pragma unroll
    for (int mi = 0; mi < 2; mi++) {
        #pragma unroll
        for (int ni = 0; ni < 4; ni++) {
            const int mbase = m0 + wm * 32 + mi * 16 + 8 * half;
            const int n     = n0 + wn * 64 + ni * 16 + row0;
            const float bv  = bias[n];
            #pragma unroll
            for (int r = 0; r < 8; r++) {
                const float vres = acc[mi][ni][r] + bv;
                if (OUT_BF16)
                    ((unsigned short*)out)[(size_t)(mbase + r) * N + n] = f2bf(vres);
                else
                    ((float*)out)[(size_t)(mbase + r) * N + n] = vres;
            }
        }
    }
}

// ---------------------------------------------------------------------------
// Per-head L2 normalize: each thread owns one contiguous 64-float segment
// (heads are contiguous 64-col blocks of the [B*S, 1024] layout).
// x / max(||x||, eps) == x * rsqrt(max(ss, eps^2))  -> single v_rsq_f32.
// ---------------------------------------------------------------------------
__global__ __launch_bounds__(256) void l2norm_bf16_kernel(
    const float* __restrict__ X, unsigned short* __restrict__ Y)
{
    const size_t seg = (size_t)blockIdx.x * 256 + threadIdx.x;
    const float4* p  = (const float4*)(X + seg * 64);
    float4 buf[16];
    float  ss = 0.0f;
    #pragma unroll
    for (int i = 0; i < 16; i++) {
        buf[i] = p[i];
        ss += buf[i].x * buf[i].x + buf[i].y * buf[i].y +
              buf[i].z * buf[i].z + buf[i].w * buf[i].w;
    }
    const float inv = __builtin_amdgcn_rsqf(fmaxf(ss, 1e-24f));
    unsigned short o[64];
    const float* fs = (const float*)buf;
    #pragma unroll
    for (int i = 0; i < 64; i++) o[i] = f2bf(fs[i] * inv);
    uint4* d = (uint4*)(Y + seg * 64);
    const uint4* s = (const uint4*)o;
    #pragma unroll
    for (int i = 0; i < 8; i++) d[i] = s[i];
}

// ---------------------------------------------------------------------------
// Fused hyperbolic attention per (batch, head, 128-row Q tile):
//   sim = Qn*Kn^T (WMMA) -> w = s - sqrt(s^2-1), s=max(sim,1) -> O += w*V (WMMA)
// The 2048x2048 sim matrix never leaves LDS/registers.
// ---------------------------------------------------------------------------
__global__ __launch_bounds__(256) void hyp_attn_kernel(
    const unsigned short* __restrict__ Qn, const unsigned short* __restrict__ Kn,
    const unsigned short* __restrict__ Vb, float* __restrict__ O,
    int S, int D)
{
    __shared__ unsigned short Qsh[128][64];   // [m][d]      bf16
    __shared__ unsigned short Ksh[64][64];    // [d][s_k]    bf16 (transposed)
    __shared__ unsigned short Vsh[64][64];    // [s_k][d]    bf16
    __shared__ unsigned short Wsh[128][64];   // [m][s_k]    bf16 weights

    const int t    = threadIdx.x;
    const int lane = t & 31;
    const int wv   = t >> 5;                  // wave owns 16 q-rows
    const int h    = blockIdx.y & 15;
    const int b    = blockIdx.y >> 4;
    const int qt   = blockIdx.x;
    const size_t rowbase = (size_t)b * S * D;

    // ---- stage Q tile (rows qt*128..+128, cols h*64..+64): pure bf16 copy
    {
        const int m = t >> 1, c = (t & 1) * 32;
        const unsigned short* gp = Qn + rowbase + (size_t)(qt * 128 + m) * D + h * 64 + c;
        void* lp = &Qsh[m][c];
#if HAVE_ASYNC_LDS
        // async DMA: INST_OFFSET applies to both global and LDS addresses
        __builtin_amdgcn_global_load_async_to_lds_b128(as_global_v4(gp), as_lds_v4(lp), 0,  0);
        __builtin_amdgcn_global_load_async_to_lds_b128(as_global_v4(gp), as_lds_v4(lp), 16, 0);
        __builtin_amdgcn_global_load_async_to_lds_b128(as_global_v4(gp), as_lds_v4(lp), 32, 0);
        __builtin_amdgcn_global_load_async_to_lds_b128(as_global_v4(gp), as_lds_v4(lp), 48, 0);
        wait_async0();
#else
        const uint4* g = (const uint4*)gp;
        uint4 t0 = g[0], t1 = g[1], t2 = g[2], t3 = g[3];
        uint4* d = (uint4*)lp;
        d[0] = t0; d[1] = t1; d[2] = t2; d[3] = t3;
#endif
    }
    __syncthreads();

    const int half = lane >> 4;
    const int row0 = lane & 15;

    // Q fragments held in registers across the whole key loop (2 k-steps of 32)
    Frag qfr[2];
    #pragma unroll
    for (int ks = 0; ks < 2; ks++) {
        const int r = wv * 16 + row0;
        qfr[ks].u[0] = *(const uint4*)&Qsh[r][ks * 32 + 8 * half];
        qfr[ks].u[1] = *(const uint4*)&Qsh[r][ks * 32 + 16 + 8 * half];
    }

    v8f oacc[4];
    for (int j = 0; j < 4; j++)
        for (int e = 0; e < 8; e++) oacc[j][e] = 0.0f;

    for (int kt = 0; kt < S / 64; kt++) {
        const int s = t >> 2, c = (t & 3) * 16;
        const size_t krow = rowbase + (size_t)(kt * 64 + s) * D + h * 64 + c;
#if HAVE_ASYNC_LDS
        // V tile: async DMA (straight copy), overlapped with K transpose below
        {
            const unsigned short* gv = Vb + krow;
            void* lv = &Vsh[s][c];
            __builtin_amdgcn_global_load_async_to_lds_b128(as_global_v4(gv), as_lds_v4(lv), 0,  0);
            __builtin_amdgcn_global_load_async_to_lds_b128(as_global_v4(gv), as_lds_v4(lv), 16, 0);
        }
#endif
        // K tile staged transposed ([d][s_k]) -- needs lane-level transpose
        {
            const unsigned short* gk = Kn + krow;
            unsigned short tmp[16];
            uint4 k0 = ((const uint4*)gk)[0];
            uint4 k1 = ((const uint4*)gk)[1];
            *(uint4*)&tmp[0] = k0;
            *(uint4*)&tmp[8] = k1;
            #pragma unroll
            for (int i = 0; i < 16; i++) Ksh[c + i][s] = tmp[i];
        }
#if HAVE_ASYNC_LDS
        wait_async0();
#else
        {
            const uint4* gv = (const uint4*)(Vb + krow);
            uint4 v0 = gv[0], v1 = gv[1];
            *(uint4*)&Vsh[s][c]     = v0;
            *(uint4*)&Vsh[s][c + 8] = v1;
        }
#endif
        __syncthreads();

        // sim tile (16 x 64 per wave) + hyperbolic weight, written to this
        // wave's private rows of Wsh (no cross-wave dependency -> no barrier)
        #pragma unroll
        for (int ni = 0; ni < 4; ni++) {
            v8f c8;
            for (int e = 0; e < 8; e++) c8[e] = 0.0f;
            #pragma unroll
            for (int ks = 0; ks < 2; ks++) {
                Frag bfr;   // B[K=d][N=s_k]: lane L holds N=L%16, d in table order
                #pragma unroll
                for (int e = 0; e < 16; e++)
                    ((unsigned short*)&bfr)[e] = Ksh[ks * 32 + (e & 7) + 8 * half + 16 * (e >> 3)][ni * 16 + row0];
                c8 = wmma_bf16(qfr[ks], bfr, c8);
            }
            #pragma unroll
            for (int r = 0; r < 8; r++) {
                const float sv = fmaxf(c8[r], 1.0f);
                // exp(-acosh(sv)) == sv - sqrt(sv*sv-1); fma keeps arg >= 0
                const float w = sv - __builtin_amdgcn_sqrtf(__builtin_fmaf(sv, sv, -1.0f));
                Wsh[wv * 16 + 8 * half + r][ni * 16 + row0] = f2bf(w);
            }
        }

        // O(16x64 per wave) += W(16x64) * V(64x64)
        #pragma unroll
        for (int ks = 0; ks < 2; ks++) {
            Frag afr;
            const int r = wv * 16 + row0;
            afr.u[0] = *(const uint4*)&Wsh[r][ks * 32 + 8 * half];
            afr.u[1] = *(const uint4*)&Wsh[r][ks * 32 + 16 + 8 * half];
            #pragma unroll
            for (int ni = 0; ni < 4; ni++) {
                Frag bfr;   // B[K=s_k][N=d]: lane L holds N=L%16, s_k in table order
                #pragma unroll
                for (int e = 0; e < 16; e++)
                    ((unsigned short*)&bfr)[e] = Vsh[ks * 32 + (e & 7) + 8 * half + 16 * (e >> 3)][ni * 16 + row0];
                oacc[ni] = wmma_bf16(afr, bfr, oacc[ni]);
            }
        }
        __syncthreads();   // protect Ksh/Vsh/Wsh before next iteration
    }

    // store O tile (fp32, merged-head [B*S, 1024] layout)
    #pragma unroll
    for (int ni = 0; ni < 4; ni++) {
        #pragma unroll
        for (int r = 0; r < 8; r++) {
            const int m = qt * 128 + wv * 16 + 8 * half + r;
            const int n = h * 64 + ni * 16 + row0;
            O[rowbase + (size_t)m * D + n] = oacc[ni][r];
        }
    }
}

// ---------------------------------------------------------------------------
extern "C" void kernel_launch(void* const* d_in, const int* in_sizes, int n_in,
                              void* d_out, int out_size, void* d_ws, size_t ws_size,
                              hipStream_t stream)
{
    const float* query = (const float*)d_in[0];
    const float* key   = (const float*)d_in[1];
    const float* value = (const float*)d_in[2];
    const float* w_q   = (const float*)d_in[3];
    const float* b_q   = (const float*)d_in[4];
    const float* w_k   = (const float*)d_in[5];
    const float* b_k   = (const float*)d_in[6];
    const float* w_v   = (const float*)d_in[7];
    const float* b_v   = (const float*)d_in[8];
    const float* w_o   = (const float*)d_in[9];
    const float* b_o   = (const float*)d_in[10];

    constexpr int B = 4, S = 2048, D = 1024;
    constexpr int M = B * S;                       // 8192
    const size_t MB = 1ull << 20;

    char* ws = (char*)d_ws;
    float*          Qf = (float*)(ws);             // 32 MB, reused as O later
    float*          Kf = (float*)(ws + 32 * MB);   // 32 MB
    unsigned short* Qn = (unsigned short*)(ws + 64 * MB);   // 16 MB
    unsigned short* Kn = (unsigned short*)(ws + 80 * MB);   // 16 MB
    unsigned short* Vb = (unsigned short*)(ws + 96 * MB);   // 16 MB
    float*          O  = (float*)(ws);             // aliases Qf (dead by then)

    const dim3 gemm_grid(D / 128, M / 128);        // (8, 64)

    gemm_bias_kernel<false><<<gemm_grid, 256, 0, stream>>>(query, w_q, b_q, Qf, M, D, D);
    gemm_bias_kernel<false><<<gemm_grid, 256, 0, stream>>>(key,   w_k, b_k, Kf, M, D, D);
    gemm_bias_kernel<true ><<<gemm_grid, 256, 0, stream>>>(value, w_v, b_v, Vb, M, D, D);

    l2norm_bf16_kernel<<<(M * 16) / 256, 256, 0, stream>>>(Qf, Qn);
    l2norm_bf16_kernel<<<(M * 16) / 256, 256, 0, stream>>>(Kf, Kn);

    hyp_attn_kernel<<<dim3(S / 128, B * 16), 256, 0, stream>>>(Qn, Kn, Vb, O, S, D);

    gemm_bias_kernel<false><<<gemm_grid, 256, 0, stream>>>(O, w_o, b_o, (float*)d_out, M, D, D);
}
